// Net_46368466928157
// MI455X (gfx1250) — matrix operand order repaired and checked by reading, not compiled, not run on hardware
//
#include <hip/hip_runtime.h>
#include <hip/hip_bf16.h>

// ---------------------------------------------------------------------------
// Sparse autoencoder forward on MI455X (gfx1250, wave32, WMMA).
// Pipeline: fp32->bf16 (pad) conversions -> GEMM1 -> GEMM2 -> topk mask ->
//           GEMM3 -> GEMM4 (fp32 out).
// GEMMs: v_wmma_f32_16x16x32_bf16 with double-buffered LDS tiles filled by
// CDNA5 async global->LDS copies (global_load_async_to_lds_b128 / ASYNCcnt).
// ---------------------------------------------------------------------------

typedef __attribute__((ext_vector_type(16))) __bf16 v16bf;
typedef __attribute__((ext_vector_type(8)))  float  v8f;

#define BATCH   16384
#define IN_DIM  784
#define K1PAD   800      // 784 padded to multiple of 32
#define INTER   1024
#define CODE    2048
#define N4PAD   832      // 784 padded to multiple of 64 (block N tile)

// ---------------------------------------------------------------------------
// fp32 -> bf16 with optional row/col zero padding.
// ---------------------------------------------------------------------------
__global__ __launch_bounds__(256)
void cvt_pad_bf16(const float* __restrict__ src, __bf16* __restrict__ dst,
                  int src_rows, int src_cols, int dst_cols, int total) {
    int i = blockIdx.x * 256 + threadIdx.x;
    if (i >= total) return;
    int r = i / dst_cols;
    int c = i - r * dst_cols;
    float v = (r < src_rows && c < src_cols) ? src[(size_t)r * src_cols + c] : 0.0f;
    dst[i] = (__bf16)v;
}

__global__ __launch_bounds__(256)
void pad_bias(const float* __restrict__ src, float* __restrict__ dst,
              int n_src, int n_dst) {
    int i = blockIdx.x * 256 + threadIdx.x;
    if (i < n_dst) dst[i] = (i < n_src) ? src[i] : 0.0f;
}

// ---------------------------------------------------------------------------
// Async global -> LDS copy of one 16B chunk per lane (CDNA5 TDM-lite path).
// lds32: LDS byte address (addrspace(3) offset); gaddr: 64-bit global address.
// ---------------------------------------------------------------------------
__device__ __forceinline__ void async_copy_b128(unsigned lds32,
                                                unsigned long long gaddr) {
    asm volatile("global_load_async_to_lds_b128 %0, %1, off"
                 :: "v"(lds32), "v"(gaddr)
                 : "memory");
}

__device__ __forceinline__ void wait_async0() {
    asm volatile("s_wait_asynccnt 0" ::: "memory");
}

// ---------------------------------------------------------------------------
// GEMM: Out[M,N] = relu(A[M,K] @ W[N,K]^T + bias[N])
// Block = 256 threads (8 waves). Block tile 128(M) x 64(N), K-step 32.
// Double-buffered LDS; async loads for tile k+1 overlap compute of tile k.
// Each wave: one 16-row M slice, four 16x16 WMMA accumulators across N=64.
// ---------------------------------------------------------------------------
template <bool OUT_BF16>
__global__ __launch_bounds__(256)
void gemm_bias_relu(const __bf16* __restrict__ A, const __bf16* __restrict__ W,
                    const float* __restrict__ bias, void* __restrict__ Out,
                    int K, int ldo, int Nstore) {
    __shared__ __align__(16) __bf16 As[2][128][40];  // +8 pad: row stride 80B
    __shared__ __align__(16) __bf16 Bs[2][64][40];

    const int tid  = threadIdx.x;
    const int wave = tid >> 5;
    const int lane = tid & 31;
    const int ln   = lane & 15;
    const int hi   = lane >> 4;           // half-wave select (K split: 0..15 | 16..31, contiguous)

    const size_t mbase = (size_t)blockIdx.y * 128;
    const int    nbase = blockIdx.x * 64;

    // Per-thread chunk coordinates for tile staging (16B chunks).
    const int ar0 = (tid + 0)   >> 2, ac0 = ((tid + 0)   & 3) << 3;  // A chunk 0
    const int ar1 = (tid + 256) >> 2, ac1 = ((tid + 256) & 3) << 3;  // A chunk 1
    const int br_ = tid >> 2,         bc_ = (tid & 3) << 3;          // B chunk

    auto issue_tile = [&](int k0, int buf) {
        async_copy_b128((unsigned)(size_t)&As[buf][ar0][ac0],
                        (unsigned long long)(A + (mbase + ar0) * (size_t)K + k0 + ac0));
        async_copy_b128((unsigned)(size_t)&As[buf][ar1][ac1],
                        (unsigned long long)(A + (mbase + ar1) * (size_t)K + k0 + ac1));
        async_copy_b128((unsigned)(size_t)&Bs[buf][br_][bc_],
                        (unsigned long long)(W + (size_t)(nbase + br_) * K + k0 + bc_));
    };

    v8f acc[4] = {};

    // Prologue: stage tile 0.
    issue_tile(0, 0);
    wait_async0();
    __syncthreads();

    const int nk = K >> 5;
    for (int k = 0; k < nk; ++k) {
        const int cur = k & 1;
        if (k + 1 < nk) issue_tile((k + 1) << 5, cur ^ 1);

        // A fragment (16x32 bf16, lane=M, K=0..15 | 16..31 contiguous).
        union F { v16bf v; uint4 u[2]; };
        F a, b[4];
        const int ar  = wave * 16 + ln;
        const int akb = hi ? 16 : 0;
        a.u[0] = *(const uint4*)&As[cur][ar][akb];
        a.u[1] = *(const uint4*)&As[cur][ar][akb + 8];

        // B fragments (32x16 bf16, lane=N, K=0..15 | 16..31 contiguous).
        const int bkb = hi ? 16 : 0;
        #pragma unroll
        for (int t = 0; t < 4; ++t) {
            const int br = t * 16 + ln;
            b[t].u[0] = *(const uint4*)&Bs[cur][br][bkb];
            b[t].u[1] = *(const uint4*)&Bs[cur][br][bkb + 8];
        }

        // Back-to-back matrix ops (XDL-friendly issue pattern).
        #pragma unroll
        for (int t = 0; t < 4; ++t) {
            acc[t] = __builtin_amdgcn_wmma_f32_16x16x32_bf16(
                false, a.v, false, b[t].v, (short)0, acc[t], false, false);
        }

        wait_async0();     // tile k+1 landed in LDS
        __syncthreads();   // all waves done reading buf `cur` / writing `cur^1`
    }

    // Epilogue: bias + relu + store. C/D layout: vgpr r -> M = r + hi*8, lane = N.
    #pragma unroll
    for (int t = 0; t < 4; ++t) {
        const int col = nbase + t * 16 + ln;
        if (col < Nstore) {
            const float bv = bias[col];
            #pragma unroll
            for (int r = 0; r < 8; ++r) {
                const size_t row = mbase + wave * 16 + r + hi * 8;
                float v = acc[t][r] + bv;
                v = fmaxf(v, 0.0f);
                if (OUT_BF16) ((__bf16*)Out)[row * (size_t)ldo + col] = (__bf16)v;
                else          ((float*)Out)[row * (size_t)ldo + col] = v;
            }
        }
    }
}

// ---------------------------------------------------------------------------
// Per-row top-64 neuron mask + top-4-of-32 stripe mask, in place on c (bf16).
// One block (256 threads) per row of 2048; each thread owns 8 consecutive
// elements (all in the same 64-wide stripe).
// ---------------------------------------------------------------------------
__global__ __launch_bounds__(256)
void topk_mask_kernel(__bf16* __restrict__ c) {
    const int row = blockIdx.x;
    const int tid = threadIdx.x;
    __shared__ float red[256];
    __shared__ float savg[32];
    __shared__ float smask[32];

    __bf16* crow = c + (size_t)row * CODE;
    float loc[8];
    #pragma unroll
    for (int i = 0; i < 8; ++i) loc[i] = (float)crow[tid * 8 + i];

    // Row max (values are >= 0 post-ReLU).
    float m = 0.0f;
    #pragma unroll
    for (int i = 0; i < 8; ++i) m = fmaxf(m, loc[i]);
    red[tid] = m;
    __syncthreads();
    for (int s = 128; s > 0; s >>= 1) {
        if (tid < s) red[tid] = fmaxf(red[tid], red[tid + s]);
        __syncthreads();
    }
    const float vmax = red[0];
    __syncthreads();

    // Binary search for the 64th-largest value: largest t with count(v>=t)>=64.
    float lo = 0.0f, hi = vmax;
    for (int it = 0; it < 24; ++it) {
        const float mid = 0.5f * (lo + hi);
        int cnt = 0;
        #pragma unroll
        for (int i = 0; i < 8; ++i) cnt += (loc[i] >= mid) ? 1 : 0;
        red[tid] = (float)cnt;
        __syncthreads();
        for (int s = 128; s > 0; s >>= 1) {
            if (tid < s) red[tid] += red[tid + s];
            __syncthreads();
        }
        const float total = red[0];
        __syncthreads();
        if (total >= 64.0f) lo = mid; else hi = mid;
    }
    const float thr = lo;

    // Neuron mask in registers; stripe partial sums.
    float p = 0.0f;
    #pragma unroll
    for (int i = 0; i < 8; ++i) {
        loc[i] = (loc[i] >= thr) ? loc[i] : 0.0f;
        p += loc[i];
    }
    red[tid] = p;
    __syncthreads();
    if (tid < 32) {  // 8 partials per stripe
        float s = 0.0f;
        #pragma unroll
        for (int j = 0; j < 8; ++j) s += red[tid * 8 + j];
        savg[tid] = s;  // mean = s/64; scaling doesn't change ranking
    }
    __syncthreads();
    if (tid == 0) {   // top-4 of 32 stripes by simple selection
        for (int s = 0; s < 32; ++s) smask[s] = 0.0f;
        for (int j = 0; j < 4; ++j) {
            int best = 0; float bv = -1.0f;
            for (int s = 0; s < 32; ++s)
                if (smask[s] == 0.0f && savg[s] > bv) { bv = savg[s]; best = s; }
            smask[best] = 1.0f;
        }
    }
    __syncthreads();

    const float sm = smask[tid >> 3];
    #pragma unroll
    for (int i = 0; i < 8; ++i)
        crow[tid * 8 + i] = (__bf16)(loc[i] * sm);
}

// ---------------------------------------------------------------------------
// Host launcher.
// ---------------------------------------------------------------------------
extern "C" void kernel_launch(void* const* d_in, const int* in_sizes, int n_in,
                              void* d_out, int out_size, void* d_ws, size_t ws_size,
                              hipStream_t stream) {
    const float* x  = (const float*)d_in[0];
    const float* W1 = (const float*)d_in[1];
    const float* b1 = (const float*)d_in[2];
    const float* W2 = (const float*)d_in[3];
    const float* b2 = (const float*)d_in[4];
    const float* W3 = (const float*)d_in[5];
    const float* b3 = (const float*)d_in[6];
    const float* W4 = (const float*)d_in[7];
    const float* b4 = (const float*)d_in[8];

    // Workspace carve-up (256B aligned slices).
    char* p = (char*)d_ws;
    auto alloc = [&](size_t bytes) -> void* {
        void* r = (void*)p;
        p += (bytes + 255) & ~(size_t)255;
        return r;
    };
    __bf16* xb  = (__bf16*)alloc((size_t)BATCH * K1PAD * 2);
    __bf16* w1b = (__bf16*)alloc((size_t)INTER * K1PAD * 2);
    __bf16* w2b = (__bf16*)alloc((size_t)CODE  * INTER * 2);
    __bf16* w3b = (__bf16*)alloc((size_t)INTER * CODE  * 2);
    __bf16* w4b = (__bf16*)alloc((size_t)N4PAD * INTER * 2);
    float*  b4p = (float*) alloc((size_t)N4PAD * 4);
    __bf16* hb  = (__bf16*)alloc((size_t)BATCH * INTER * 2);
    __bf16* cb  = (__bf16*)alloc((size_t)BATCH * CODE  * 2);
    __bf16* db  = (__bf16*)alloc((size_t)BATCH * INTER * 2);

    // --- conversions ---
    {
        int t = BATCH * K1PAD;
        cvt_pad_bf16<<<(t + 255) / 256, 256, 0, stream>>>(x, xb, BATCH, IN_DIM, K1PAD, t);
    }
    {
        int t = INTER * K1PAD;
        cvt_pad_bf16<<<(t + 255) / 256, 256, 0, stream>>>(W1, w1b, INTER, IN_DIM, K1PAD, t);
    }
    {
        int t = CODE * INTER;
        cvt_pad_bf16<<<(t + 255) / 256, 256, 0, stream>>>(W2, w2b, CODE, INTER, INTER, t);
    }
    {
        int t = INTER * CODE;
        cvt_pad_bf16<<<(t + 255) / 256, 256, 0, stream>>>(W3, w3b, INTER, CODE, CODE, t);
    }
    {
        int t = N4PAD * INTER;
        cvt_pad_bf16<<<(t + 255) / 256, 256, 0, stream>>>(W4, w4b, IN_DIM, INTER, INTER, t);
    }
    pad_bias<<<(N4PAD + 255) / 256, 256, 0, stream>>>(b4, b4p, IN_DIM, N4PAD);

    // --- GEMM1: h = relu(x @ W1^T + b1)   [16384,800]x[1024,800] ---
    {
        dim3 g(INTER / 64, BATCH / 128);
        gemm_bias_relu<true><<<g, 256, 0, stream>>>(xb, w1b, b1, hb, K1PAD, INTER, INTER);
    }
    // --- GEMM2: c = relu(h @ W2^T + b2)   [16384,1024]x[2048,1024] ---
    {
        dim3 g(CODE / 64, BATCH / 128);
        gemm_bias_relu<true><<<g, 256, 0, stream>>>(hb, w2b, b2, cb, INTER, CODE, CODE);
    }
    // --- top-k neuron + stripe masking (in place on c) ---
    topk_mask_kernel<<<BATCH, 256, 0, stream>>>(cb);
    // --- GEMM3: d = relu(c @ W3^T + b3)   [16384,2048]x[1024,2048] ---
    {
        dim3 g(INTER / 64, BATCH / 128);
        gemm_bias_relu<true><<<g, 256, 0, stream>>>(cb, w3b, b3, db, CODE, INTER, INTER);
    }
    // --- GEMM4: out = relu(d @ W4^T + b4) [16384,1024]x[832,1024], store 784 ---
    {
        dim3 g(N4PAD / 64, BATCH / 128);
        gemm_bias_relu<false><<<g, 256, 0, stream>>>(db, w4b, b4p, d_out, INTER, IN_DIM, IN_DIM);
    }
}
